// gen_new_4750233829504
// MI455X (gfx1250) — matrix operand order repaired and checked by reading, not compiled, not run on hardware
//
#include <hip/hip_runtime.h>

typedef _Float16 h8   __attribute__((ext_vector_type(8)));
typedef _Float16 v16h __attribute__((ext_vector_type(16)));
typedef float    v8f  __attribute__((ext_vector_type(8)));

#define G     64
#define GP    (G * G * G)      // 262144 latent grid points
#define DD    256              // output volume edge
#define HID   256              // hidden width
#define MB    64               // grid points (rows) per MLP block
#define LDSTR 264              // padded LDS row stride in halves (breaks bank conflicts)

// ---------------------------------------------------------------------------
// Prep: fold z into layer-1 bias, convert W2/W3 to f16 for WMMA.
// ---------------------------------------------------------------------------
__global__ __launch_bounds__(256) void prep_kernel(
    const float* __restrict__ z, const float* __restrict__ W1,
    const float* __restrict__ b1, const float* __restrict__ W2,
    const float* __restrict__ W3, float* __restrict__ c1,
    _Float16* __restrict__ W2h, _Float16* __restrict__ W3h) {
  const int j = threadIdx.x;  // 0..255 : one hidden neuron per thread
  c1[j] = b1[j] + z[0] * W1[j * 6 + 0] + z[1] * W1[j * 6 + 1] + z[2] * W1[j * 6 + 2];
  for (int k = 0; k < HID; ++k) {
    W2h[j * HID + k] = (_Float16)W2[j * HID + k];
    W3h[j * HID + k] = (_Float16)W3[j * HID + k];
  }
}

// ---------------------------------------------------------------------------
// Fused MLP: h1 (VALU) -> WMMA layer2 -> WMMA layer3 -> layer4 + tanh (VALU)
// Block = 256 threads = 8 wave32s, handles MB=64 grid points.
// Loop nest keeps B (weights) register-resident per N-tile; A streams from LDS
// so each block reads W2/W3 exactly once (minimal L2 traffic).
// ---------------------------------------------------------------------------
__global__ __launch_bounds__(256) void mlp_kernel(
    const float* __restrict__ W1, const float* __restrict__ c1,
    const _Float16* __restrict__ W2h, const float* __restrict__ b2,
    const _Float16* __restrict__ W3h, const float* __restrict__ b3,
    const float* __restrict__ W4, const float* __restrict__ b4,
    float* __restrict__ flow) {
  __shared__ _Float16 bufA[MB * LDSTR];
  __shared__ _Float16 bufB[MB * LDSTR];

  const int tid = threadIdx.x;
  const int r0  = blockIdx.x * MB;

  // ---- layer 1: h1 = relu(c1 + gx*w3 + gy*w4 + gz*w5), col fixed per thread
  {
    const int   col = tid;
    const float wgx = W1[col * 6 + 3];
    const float wgy = W1[col * 6 + 4];
    const float wgz = W1[col * 6 + 5];
    const float cc  = c1[col];
    const float inv = 1.0f / 63.0f;  // linspace(0,1,64)
    for (int row = 0; row < MB; ++row) {
      const int n  = r0 + row;
      const float gx = (float)(n >> 12) * inv;
      const float gy = (float)((n >> 6) & 63) * inv;
      const float gz = (float)(n & 63) * inv;
      const float v  = cc + gx * wgx + gy * wgy + gz * wgz;
      bufA[row * LDSTR + col] = (_Float16)fmaxf(v, 0.0f);
    }
  }
  __syncthreads();

  // ---- layers 2 & 3: 16x16 tiles, K=256 -> 8x v_wmma_f32_16x16x32_f16 each
  const int lane = tid & 31;
  const int wave = tid >> 5;
  const int lrow = lane & 15;  // M (A) or N (B/C/D) index within tile
  const int lhi  = lane >> 4;  // lane-group select

  for (int layer = 0; layer < 2; ++layer) {
    const _Float16* __restrict__ Wg  = (layer == 0) ? W2h : W3h;
    const float*    __restrict__ bia = (layer == 0) ? b2  : b3;
    const _Float16* src = (layer == 0) ? bufA : bufB;
    _Float16*       dst = (layer == 0) ? bufB : bufA;

    // Each wave owns 2 N-tiles (8 waves x 2 = 16 = all 256 output columns).
    for (int t = 0; t < 2; ++t) {
      const int nt = wave * 2 + t;

      // Load ALL K-steps of this N-tile's B fragment into registers (once).
      // B 32x16 f16 layout: e0..15 -> K = K0 + lhi*16 + e (contiguous halves)
      const _Float16* brow = Wg + (nt * 16 + lrow) * HID + lhi * 16;
      v16h bf[8];
      #pragma unroll
      for (int kk = 0; kk < 8; ++kk) {
        h8 blo = *(const h8*)(brow + kk * 32);
        h8 bhi = *(const h8*)(brow + kk * 32 + 8);
        #pragma unroll
        for (int e = 0; e < 8; ++e) { bf[kk][e] = blo[e]; bf[kk][e + 8] = bhi[e]; }
      }

      const int   col = nt * 16 + lrow;
      const float bv  = bia[col];

      // Stream the 4 M-tiles of activations from LDS against resident B.
      #pragma unroll
      for (int mt = 0; mt < 4; ++mt) {
        v8f acc = {};
        // A 16x32 f16 layout: e0..15 -> K0+lhi*16+e (contiguous halves)
        const _Float16* arow = src + (mt * 16 + lrow) * LDSTR + lhi * 16;
        #pragma unroll
        for (int kk = 0; kk < 8; ++kk) {
          h8 alo = *(const h8*)(arow + kk * 32);
          h8 ahi = *(const h8*)(arow + kk * 32 + 8);
          v16h a;
          #pragma unroll
          for (int e = 0; e < 8; ++e) { a[e] = alo[e]; a[e + 8] = ahi[e]; }
          acc = __builtin_amdgcn_wmma_f32_16x16x32_f16(
              false, a, false, bf[kk], (short)0, acc, false, false);
        }
        // C/D layout: N = col (per lane), M = mt*16 + lhi*8 + v
        #pragma unroll
        for (int v = 0; v < 8; ++v) {
          const int mr = mt * 16 + lhi * 8 + v;
          dst[mr * LDSTR + col] = (_Float16)fmaxf(acc[v] + bv, 0.0f);
        }
      }
    }
    __syncthreads();
  }

  // ---- layer 4: flow_c = tanh(h3 . W4[c] + b4[c]) * 8   (3 outputs / row)
  if (tid < MB * 3) {
    const int row = tid / 3;
    const int c   = tid % 3;
    float sum = b4[c];
    const h8*    hrow = (const h8*)(bufA + row * LDSTR);  // 16B-aligned rows
    const float* w4   = W4 + c * HID;
    for (int k8 = 0; k8 < HID / 8; ++k8) {
      h8 hv = hrow[k8];
      #pragma unroll
      for (int e = 0; e < 8; ++e) sum += (float)hv[e] * w4[k8 * 8 + e];
    }
    flow[c * GP + r0 + row] = tanhf(sum) * 8.0f;
  }
}

// ---------------------------------------------------------------------------
// Warp: 4x nearest-upsampled flow + base coords, zero-padded trilinear sample
// of img [2,256,256,256]. One thread per voxel, both channels. Non-temporal
// output stores keep img resident in the 192 MB L2.
// ---------------------------------------------------------------------------
__global__ __launch_bounds__(256) void warp_sample_kernel(
    const float* __restrict__ img, const float* __restrict__ flow,
    float* __restrict__ out) {
  const int idx = blockIdx.x * 256 + threadIdx.x;  // (d*256+h)*256+w
  const int w = idx & 255;
  const int h = (idx >> 8) & 255;
  const int d = idx >> 16;
  const int n4 = (((d >> 2) << 6) + (h >> 2)) * 64 + (w >> 2);

  const float cd = (float)d + flow[n4];
  const float ch = (float)h + flow[GP + n4];
  const float cw = (float)w + flow[2 * GP + n4];

  const float d0f = floorf(cd), h0f = floorf(ch), w0f = floorf(cw);
  const float fd = cd - d0f, fh = ch - h0f, fw = cw - w0f;
  const int d0 = (int)d0f, h0 = (int)h0f, w0 = (int)w0f;

  float acc0 = 0.0f, acc1 = 0.0f;
  const long plane = (long)DD * DD * DD;

  #pragma unroll
  for (int td = 0; td < 2; ++td) {
    const int   di = d0 + td;
    const float wd = td ? fd : 1.0f - fd;
    const bool  vd = (di >= 0) && (di < DD);
    const int   dc = min(max(di, 0), DD - 1);
    #pragma unroll
    for (int th_ = 0; th_ < 2; ++th_) {
      const int   hi = h0 + th_;
      const float wh = th_ ? fh : 1.0f - fh;
      const bool  vh = (hi >= 0) && (hi < DD);
      const int   hc = min(max(hi, 0), DD - 1);
      #pragma unroll
      for (int tw = 0; tw < 2; ++tw) {
        const int   wi  = w0 + tw;
        const float wwt = tw ? fw : 1.0f - fw;
        const bool  vw  = (wi >= 0) && (wi < DD);
        const int   wc  = min(max(wi, 0), DD - 1);
        const float wt  = (vd && vh && vw) ? wd * wh * wwt : 0.0f;
        const long  off = ((long)dc * DD + hc) * DD + wc;
        acc0 += wt * img[off];
        acc1 += wt * img[plane + off];
      }
    }
  }
  __builtin_nontemporal_store(acc0, out + idx);
  __builtin_nontemporal_store(acc1, out + plane + idx);
}

// ---------------------------------------------------------------------------
extern "C" void kernel_launch(void* const* d_in, const int* in_sizes, int n_in,
                              void* d_out, int out_size, void* d_ws, size_t ws_size,
                              hipStream_t stream) {
  const float* z   = (const float*)d_in[0];
  const float* img = (const float*)d_in[1];
  const float* W1  = (const float*)d_in[2];
  const float* b1  = (const float*)d_in[3];
  const float* W2  = (const float*)d_in[4];
  const float* b2  = (const float*)d_in[5];
  const float* W3  = (const float*)d_in[6];
  const float* b3  = (const float*)d_in[7];
  const float* W4  = (const float*)d_in[8];
  const float* b4  = (const float*)d_in[9];
  float* out = (float*)d_out;

  // Workspace layout: flow[3*GP] f32 | W2h f16 | W3h f16 | c1[256] f32  (~3.4 MB)
  char* ws = (char*)d_ws;
  float*    flow = (float*)ws;
  _Float16* W2h  = (_Float16*)(ws + (size_t)3 * GP * sizeof(float));
  _Float16* W3h  = W2h + (size_t)HID * HID;
  float*    c1   = (float*)(W3h + (size_t)HID * HID);

  prep_kernel<<<1, 256, 0, stream>>>(z, W1, b1, W2, W3, c1, W2h, W3h);
  mlp_kernel<<<GP / MB, 256, 0, stream>>>(W1, c1, W2h, b2, W3h, b3, W4, b4, flow);
  warp_sample_kernel<<<(DD * DD * DD) / 256, 256, 0, stream>>>(img, flow, out);
}
